// ScaleDotProductAttention_55929064128806
// MI455X (gfx1250) — matrix operand order repaired and checked by reading, not compiled, not run on hardware
//
#include <hip/hip_runtime.h>
#include <hip/hip_bf16.h>

// ---- constants (fixed by reference) ----
#define BB 4
#define SS 4096
#define EE 512
#define AA 64
#define NT (SS/16)      // 256 query tiles per batch
#define NC (SS/32)      // 128 key chunks per batch

typedef __attribute__((ext_vector_type(16))) __bf16 v16bf;
typedef __attribute__((ext_vector_type(8)))  __bf16 v8bf;
typedef __attribute__((ext_vector_type(8)))  float  v8f;

static __device__ __forceinline__ v8f wmma_bf16(v16bf a, v16bf b, v8f c) {
    return __builtin_amdgcn_wmma_f32_16x16x32_bf16(false, a, false, b, (short)0, c, false, false);
}

// ------------------------------------------------------------------
// Stage weights: Wt[m][col][k] = W_m[k][col] (bf16), fold 1/8 into Wq
// ------------------------------------------------------------------
__global__ void wt_kernel(const float* __restrict__ Wq, const float* __restrict__ Wk,
                          const float* __restrict__ Wv, __bf16* __restrict__ Wt) {
    int idx = blockIdx.x * blockDim.x + threadIdx.x;
    if (idx >= 3 * AA * EE) return;
    int m   = idx / (AA * EE);
    int rem = idx % (AA * EE);
    int col = rem / EE;
    int k   = rem % EE;
    const float* W = (m == 0) ? Wq : ((m == 1) ? Wk : Wv);
    float v = W[k * AA + col];
    if (m == 0) v *= 0.125f;         // fold 1/sqrt(A) into q projection
    Wt[idx] = (__bf16)v;
}

// ------------------------------------------------------------------
// Fused QKV projection. One wave per (matrix, 16-row tile).
// q,k -> row-major bf16 ; v -> transposed vT[b][a][s] bf16
// ------------------------------------------------------------------
__global__ __launch_bounds__(128)
void proj_kernel(const float* __restrict__ query, const float* __restrict__ key,
                 const float* __restrict__ value, const __bf16* __restrict__ Wt,
                 const float* __restrict__ bq, const float* __restrict__ bk,
                 const float* __restrict__ bv,
                 __bf16* __restrict__ qbuf, __bf16* __restrict__ kbuf,
                 __bf16* __restrict__ vT) {
    const int lane = threadIdx.x & 31;
    const int half = lane >> 4;
    const int l16  = lane & 15;
    const int widx = blockIdx.x * 4 + (threadIdx.x >> 5);
    const int m    = widx / (BB * NT);     // 0=q,1=k,2=v
    const int tile = widx % (BB * NT);
    const float* x    = (m == 0) ? query : ((m == 1) ? key : value);
    const float* bias = (m == 0) ? bq    : ((m == 1) ? bk  : bv);
    const __bf16* W   = Wt + (size_t)m * AA * EE;
    const int row0 = tile * 16;

    v8f acc[4];
#pragma unroll
    for (int n = 0; n < 4; ++n) acc[n] = (v8f){};

    const float* xrow = x + (size_t)(row0 + l16) * EE;   // A-layout: M = lane&15
#pragma unroll 2
    for (int kk = 0; kk < 16; ++kk) {
        const int kb = kk * 32 + half * 16;
        const float* p0 = xrow + kb;
        v16bf bop[4];
        const __bf16* wb = W + kk * 32 + half * 16;
#pragma unroll
        for (int n = 0; n < 4; ++n)
            bop[n] = *(const v16bf*)(wb + (size_t)(n * 16 + l16) * EE);
        v16bf a;
#pragma unroll
        for (int e = 0; e < 8; ++e) a[e]     = (__bf16)p0[e];
#pragma unroll
        for (int e = 0; e < 8; ++e) a[8 + e] = (__bf16)p0[8 + e];
        // keep all operand loads above the WMMA block -> one clause,
        // one wait, four back-to-back WMMAs
        __builtin_amdgcn_sched_barrier(0);
#pragma unroll
        for (int n = 0; n < 4; ++n)
            acc[n] = wmma_bf16(a, bop[n], acc[n]);
    }

    const float bscale = (m == 0) ? 0.125f : 1.0f;
    if (m < 2) {
        __bf16* outb = (m == 0) ? qbuf : kbuf;
#pragma unroll
        for (int n = 0; n < 4; ++n) {
            const float bbv = bias[n * 16 + l16] * bscale;
#pragma unroll
            for (int r = 0; r < 8; ++r) {
                const int row = row0 + r + 8 * half;
                outb[(size_t)row * AA + n * 16 + l16] = (__bf16)(acc[n][r] + bbv);
            }
        }
    } else {
        const int b  = row0 >> 12;        // /SS
        const int s0 = row0 & (SS - 1);
#pragma unroll
        for (int n = 0; n < 4; ++n) {
            const float bbv = bias[n * 16 + l16];
            v8bf pk;
#pragma unroll
            for (int r = 0; r < 8; ++r) pk[r] = (__bf16)(acc[n][r] + bbv);
            *(v8bf*)(vT + ((size_t)(b * AA + n * 16 + l16) * SS) + s0 + 8 * half) = pk;
        }
    }
}

// ------------------------------------------------------------------
// Suffix sums of v at 32-key granularity (for multiplicative mask)
// ------------------------------------------------------------------
__global__ void suff_partial(const __bf16* __restrict__ vT, float* __restrict__ partial) {
    const int b  = blockIdx.x >> 7;
    const int jj = blockIdx.x & 127;
    const int a  = threadIdx.x;           // 64 threads
    const __bf16* p = vT + (size_t)(b * AA + a) * SS + jj * 32;
    float s = 0.f;
#pragma unroll
    for (int t = 0; t < 32; ++t) s += (float)p[t];
    partial[(size_t)(b * NC + jj) * AA + a] = s;
}

__global__ void suff_scan(const float* __restrict__ partial, float* __restrict__ suff) {
    const int b = blockIdx.x;
    const int a = threadIdx.x;            // 64 threads
    float s = 0.f;
    suff[(size_t)(b * (NC + 1) + NC) * AA + a] = 0.f;
    for (int jj = NC - 1; jj >= 0; --jj) {
        s += partial[(size_t)(b * NC + jj) * AA + a];
        suff[(size_t)(b * (NC + 1) + jj) * AA + a] = s;
    }
}

// ------------------------------------------------------------------
// Flash attention helpers
// ------------------------------------------------------------------
static __device__ __forceinline__ void loadK(v16bf (&kb)[4], const __bf16* kbase,
                                             int t0, int half, int l16) {
#pragma unroll
    for (int st = 0; st < 2; ++st) {
        const __bf16* krow = kbase + (size_t)(t0 + st * 16 + l16) * AA;
        kb[st * 2 + 0] = *(const v16bf*)(krow + half * 16);
        kb[st * 2 + 1] = *(const v16bf*)(krow + 32 + half * 16);
    }
}

// One 32-key step. kcur was loaded one iteration ago (latency covered);
// loads the NEXT step's K into knxt before computing.
static __device__ __forceinline__ void flash_step(
        int i, int t0, bool more, int maskflag, int half, int l16,
        const __bf16* __restrict__ kbase, const __bf16* __restrict__ vtbase,
        v16bf (&kcur)[4], v16bf (&knxt)[4], const v16bf (&qa)[2],
        float (&mrow)[8], float (&lrow)[8], v8f (&acc)[4], __bf16* myp)
{
    // V operands for current step: consumed only after the softmax section
    v16bf vb[4];
#pragma unroll
    for (int n = 0; n < 4; ++n)
        vb[n] = *(const v16bf*)(vtbase + (size_t)(n * 16 + l16) * SS + t0 + half * 16);
    // K operands for NEXT step (register double-buffer)
    if (more) loadK(knxt, kbase, t0 + 32, half, l16);
    if (more) __builtin_prefetch(vtbase + (size_t)l16 * SS + t0 + 32, 0, 0);
    __builtin_amdgcn_sched_barrier(0);

    // ---- scores: Q (16x64) x K^T (64x32) -> two 16x16 tiles ----
    v8f sc[2];
#pragma unroll
    for (int st = 0; st < 2; ++st) {
        v8f s = (v8f){};
        s = wmma_bf16(qa[0], kcur[st * 2 + 0], s);
        s = wmma_bf16(qa[1], kcur[st * 2 + 1], s);
        sc[st] = s;
    }
    // ---- multiplicative tril mask: masked scores become exactly 0 ----
    if (maskflag) {
#pragma unroll
        for (int st = 0; st < 2; ++st) {
            const int t = t0 + st * 16 + l16;
#pragma unroll
            for (int r = 0; r < 8; ++r) {
                const int sg = i * 16 + r + 8 * half;
                if (t > sg) sc[st][r] = 0.f;
            }
        }
    }
    // ---- online softmax (rows live in 16-lane halves) ----
    float corr[8];
#pragma unroll
    for (int r = 0; r < 8; ++r) {
        float v = fmaxf(sc[0][r], sc[1][r]);
        v = fmaxf(v, __shfl_xor(v, 1));
        v = fmaxf(v, __shfl_xor(v, 2));
        v = fmaxf(v, __shfl_xor(v, 4));
        v = fmaxf(v, __shfl_xor(v, 8));
        const float mn = fmaxf(mrow[r], v);
        corr[r] = __expf(mrow[r] - mn);
        mrow[r] = mn;
    }
#pragma unroll
    for (int r = 0; r < 8; ++r) {
        const float p0 = __expf(sc[0][r] - mrow[r]);
        const float p1 = __expf(sc[1][r] - mrow[r]);
        sc[0][r] = p0; sc[1][r] = p1;
        float t = p0 + p1;
        t += __shfl_xor(t, 1);
        t += __shfl_xor(t, 2);
        t += __shfl_xor(t, 4);
        t += __shfl_xor(t, 8);
        lrow[r] = lrow[r] * corr[r] + t;
    }
#pragma unroll
    for (int n = 0; n < 4; ++n)
#pragma unroll
        for (int r = 0; r < 8; ++r) acc[n][r] *= corr[r];

    // ---- P: D-layout -> A-layout via LDS ----
#pragma unroll
    for (int st = 0; st < 2; ++st)
#pragma unroll
        for (int r = 0; r < 8; ++r)
            myp[(r + 8 * half) * 40 + st * 16 + l16] = (__bf16)sc[st][r];
    asm volatile("s_wait_dscnt 0" ::: "memory");
    v16bf pa;
    {
        const __bf16* prow = myp + l16 * 40;
        v8bf c0 = *(const v8bf*)(prow + 16 * half);
        v8bf c1 = *(const v8bf*)(prow + 16 * half + 8);
#pragma unroll
        for (int e = 0; e < 8; ++e) { pa[e] = c0[e]; pa[8 + e] = c1[e]; }
    }
    // ---- PV: P (16x32) x V (32x64) ----
#pragma unroll
    for (int n = 0; n < 4; ++n)
        acc[n] = wmma_bf16(pa, vb[n], acc[n]);
}

// One 16-row query tile with exact multiplicative tril-mask softmax.
static __device__ __forceinline__ void flash_one_tile(
        int b, int i, int maskflag, int half, int l16,
        const __bf16* __restrict__ qbuf, const __bf16* __restrict__ kbuf,
        const __bf16* __restrict__ vT,   const float* __restrict__ suff,
        float* __restrict__ out, __bf16* myp)
{
    const int nsteps = maskflag ? (i / 2 + 1) : NC;
    const __bf16* kbase  = kbuf + (size_t)b * SS * AA;
    const __bf16* vtbase = vT   + (size_t)b * AA * SS;

    // Q A-operands (K-dim 64 = 2 ops of 32)
    const __bf16* qrow = qbuf + (size_t)(b * SS + i * 16 + l16) * AA;
    v16bf qa[2];
#pragma unroll
    for (int kk = 0; kk < 2; ++kk) {
        const int base = kk * 32 + half * 16;
        v8bf c0 = *(const v8bf*)(qrow + base);
        v8bf c1 = *(const v8bf*)(qrow + base + 8);
#pragma unroll
        for (int e = 0; e < 8; ++e) { qa[kk][e] = c0[e]; qa[kk][8 + e] = c1[e]; }
    }

    float mrow[8], lrow[8];
    v8f acc[4];
#pragma unroll
    for (int r = 0; r < 8; ++r) { mrow[r] = 0.f; lrow[r] = 0.f; }
#pragma unroll
    for (int n = 0; n < 4; ++n) acc[n] = (v8f){};

    // ping-pong K buffers, unrolled by 2 so there are no register moves
    v16bf k0[4], k1[4];
    loadK(k0, kbase, 0, half, l16);
    int j2 = 0;
    for (;;) {
        bool more = (j2 + 1 < nsteps);
        flash_step(i, j2 * 32, more, maskflag, half, l16, kbase, vtbase,
                   k0, k1, qa, mrow, lrow, acc, myp);
        if (!more) break;
        ++j2;
        more = (j2 + 1 < nsteps);
        flash_step(i, j2 * 32, more, maskflag, half, l16, kbase, vtbase,
                   k1, k0, qa, mrow, lrow, acc, myp);
        if (!more) break;
        ++j2;
    }

    // ---- epilogue: future masked-zero contributions + normalize ----
    const float cnt = maskflag ? (float)(SS - 32 * nsteps) : 0.f;
    float er[8];
#pragma unroll
    for (int r = 0; r < 8; ++r) er[r] = __expf(-mrow[r]);
#pragma unroll
    for (int n = 0; n < 4; ++n) {
        const float sv = suff[(size_t)(b * (NC + 1) + nsteps) * AA + n * 16 + l16];
#pragma unroll
        for (int r = 0; r < 8; ++r) {
            const float denom = lrow[r] + cnt * er[r];
            const float val = (acc[n][r] + er[r] * sv) / denom;
            out[(size_t)(b * SS + i * 16 + r + 8 * half) * AA + n * 16 + l16] = val;
        }
    }
}

// ------------------------------------------------------------------
// Wave handles query tiles j and NT-1-j (sequentially) so every wave
// does a uniform ~NC+1 steps -> no triangular load imbalance.
// ------------------------------------------------------------------
__global__ __launch_bounds__(128)
void flash_kernel(const __bf16* __restrict__ qbuf, const __bf16* __restrict__ kbuf,
                  const __bf16* __restrict__ vT,   const float* __restrict__ suff,
                  const int* __restrict__ maskp,   float* __restrict__ out) {
    __shared__ __bf16 plds[4][16 * 40];   // per-wave 16x32 P tile, stride 40 (80B)
    const int lane = threadIdx.x & 31;
    const int half = lane >> 4;
    const int l16  = lane & 15;
    const int w    = threadIdx.x >> 5;
    const int widx = blockIdx.x * 4 + w;  // 0..511
    const int b = widx >> 7;              // 128 pairs per batch
    const int j = widx & 127;
    const int maskflag = *maskp;
    __bf16* myp = plds[w];

    flash_one_tile(b, j,          maskflag, half, l16, qbuf, kbuf, vT, suff, out, myp);
    flash_one_tile(b, NT - 1 - j, maskflag, half, l16, qbuf, kbuf, vT, suff, out, myp);
}

// ------------------------------------------------------------------
extern "C" void kernel_launch(void* const* d_in, const int* in_sizes, int n_in,
                              void* d_out, int out_size, void* d_ws, size_t ws_size,
                              hipStream_t stream) {
    (void)in_sizes; (void)n_in; (void)out_size; (void)ws_size;
    const float* query = (const float*)d_in[0];
    const float* key   = (const float*)d_in[1];
    const float* value = (const float*)d_in[2];
    const float* Wq    = (const float*)d_in[3];
    const float* bq    = (const float*)d_in[4];
    const float* Wk    = (const float*)d_in[5];
    const float* bk    = (const float*)d_in[6];
    const float* Wv    = (const float*)d_in[7];
    const float* bv    = (const float*)d_in[8];
    const int*   maskp = (const int*)d_in[9];
    float* out = (float*)d_out;

    char* ws = (char*)d_ws;
    size_t off = 0;
    __bf16* Wt   = (__bf16*)(ws + off); off += (size_t)3 * AA * EE * 2;        // 196608
    __bf16* qbuf = (__bf16*)(ws + off); off += (size_t)BB * SS * AA * 2;       // 2 MB
    __bf16* kbuf = (__bf16*)(ws + off); off += (size_t)BB * SS * AA * 2;       // 2 MB
    __bf16* vT   = (__bf16*)(ws + off); off += (size_t)BB * SS * AA * 2;       // 2 MB
    float* partial = (float*)(ws + off); off += (size_t)BB * NC * AA * 4;      // 128 KB
    float* suffv   = (float*)(ws + off); off += (size_t)BB * (NC + 1) * AA * 4;// 132 KB

    wt_kernel<<<(3 * AA * EE + 255) / 256, 256, 0, stream>>>(Wq, Wk, Wv, Wt);
    proj_kernel<<<(3 * BB * NT) / 4, 128, 0, stream>>>(query, key, value, Wt,
                                                       bq, bk, bv, qbuf, kbuf, vT);
    suff_partial<<<BB * NC, AA, 0, stream>>>(vT, partial);
    suff_scan<<<BB, AA, 0, stream>>>(partial, suffv);
    flash_kernel<<<(BB * 128) / 4, 128, 0, stream>>>(qbuf, kbuf, vT, suffv, maskp, out);
}
